// CustomTripletMarginLoss_64321430225298
// MI455X (gfx1250) — compile-verified, matrix operand order checked by reading, not run on hardware
//
#include <hip/hip_runtime.h>
#include <hip/hip_bf16.h>

// Problem constants (match reference)
#define B_N     8192
#define D_K     128
#define EPSV    1e-6f
#define MARGINV 1.0f
#define RATIO_HARD 0.8f
#define BIGF    3.0e38f

typedef __attribute__((ext_vector_type(16))) __bf16 bf16x16;
typedef __attribute__((ext_vector_type(8)))  float  v8f;
typedef unsigned int uint32x4 __attribute__((ext_vector_type(4)));
typedef unsigned int uint32x8 __attribute__((ext_vector_type(8)));

struct U4 { unsigned int x, y, z, w; };
struct U2 { unsigned int x, y; };

union Frag {
    bf16x16 v;
    struct { U4 lo, hi; } q;
};

__device__ __forceinline__ unsigned short f2bf(float f) {
    unsigned int x = __float_as_uint(f);
    x += 0x7FFFu + ((x >> 16) & 1u);   // round-to-nearest-even
    return (unsigned short)(x >> 16);
}

__device__ __forceinline__ unsigned int ihash(unsigned int x) {
    x ^= x >> 16; x *= 0x7FEB352Du;
    x ^= x >> 15; x *= 0x846CA68Bu;
    x ^= x >> 16; return x;
}

// ---------------------------------------------------------------------------
// TDM: async 64-row x 128-col bf16 panel copy, global -> LDS.
// D# group0 (4 SGPRs) + group1 (8 SGPRs), 2D tensor, data_size = 2 bytes.
// ---------------------------------------------------------------------------
__device__ __forceinline__ void tdm_panel_load(const unsigned short* gsrc,
                                               unsigned lds_off) {
    unsigned long long ga = (unsigned long long)(uintptr_t)gsrc;
    uint32x4 g0;
    g0[0] = 1u;                                   // count=1, user mode
    g0[1] = lds_off;                              // lds_addr (bytes)
    g0[2] = (unsigned)ga;                         // global_addr[31:0]
    g0[3] = (unsigned)((ga >> 32) & 0x01FFFFFFull) | 0x80000000u; // addr[56:32] | type=2
    uint32x8 g1;
    g1[0] = 0x00010000u;                          // data_size=1 (2B); mask=0; no pad/iter
    g1[1] = (unsigned)D_K << 16;                  // tensor_dim0 = 128 (abar_addr=0)
    g1[2] = ((unsigned)B_N & 0xFFFFu) << 16;      // tensor_dim1 lo16 (dim0 hi16 = 0)
    g1[3] = (unsigned)D_K << 16;                  // tile_dim0 = 128 (dim1 hi16 = 0)
    g1[4] = 64u;                                  // tile_dim1 = 64 rows (tile_dim2=0)
    g1[5] = (unsigned)D_K;                        // tensor_dim0_stride = 128
    g1[6] = 0u;
    g1[7] = 0u;
    asm volatile("tensor_load_to_lds %0, %1" :: "s"(g0), "s"(g1) : "memory");
}

// ---------------------------------------------------------------------------
// Kernel 1: per-row stats + bf16 conversion (e and -2e copies).
// One wave (32 lanes) per row.
// ---------------------------------------------------------------------------
__global__ __launch_bounds__(256)
void prep_kernel(const float* __restrict__ e, const float* __restrict__ pe,
                 unsigned short* __restrict__ ebf,
                 unsigned short* __restrict__ am2,
                 float* __restrict__ sq, float* __restrict__ srow,
                 float* __restrict__ thr2) {
    const int wv   = threadIdx.x >> 5;
    const int lane = threadIdx.x & 31;
    const int row  = blockIdx.x * 8 + wv;

    const float4 ev = *(const float4*)(e  + row * D_K + lane * 4);
    const float4 pv = *(const float4*)(pe + row * D_K + lane * 4);

    float s = ev.x + ev.y + ev.z + ev.w;
    float q = ev.x*ev.x + ev.y*ev.y + ev.z*ev.z + ev.w*ev.w;
    float dx = ev.x - pv.x + EPSV, dy = ev.y - pv.y + EPSV;
    float dz = ev.z - pv.z + EPSV, dw = ev.w - pv.w + EPSV;
    float pd = dx*dx + dy*dy + dz*dz + dw*dw;

    #pragma unroll
    for (int off = 16; off >= 1; off >>= 1) {
        s  += __shfl_xor(s,  off, 32);
        q  += __shfl_xor(q,  off, 32);
        pd += __shfl_xor(pd, off, 32);
    }

    // bf16 stores: 4 elems per lane = 8 bytes each array
    U2 u2;
    u2.x = (unsigned)f2bf(ev.x) | ((unsigned)f2bf(ev.y) << 16);
    u2.y = (unsigned)f2bf(ev.z) | ((unsigned)f2bf(ev.w) << 16);
    *((U2*)(ebf + row * D_K) + lane) = u2;
    // -2*e is exact in bf16 (sign flip + exponent increment)
    U2 m2;
    m2.x = (unsigned)f2bf(-2.0f * ev.x) | ((unsigned)f2bf(-2.0f * ev.y) << 16);
    m2.y = (unsigned)f2bf(-2.0f * ev.z) | ((unsigned)f2bf(-2.0f * ev.w) << 16);
    *((U2*)(am2 + row * D_K) + lane) = m2;

    if (lane == 0) {
        sq[row]   = q;
        srow[row] = s;
        float u = (float)(ihash((unsigned)row ^ 0xC0FFEEu) >> 8) * (1.0f / 16777216.0f);
        float th = (u < RATIO_HARD) ? 0.0f : MARGINV;  // strict-hard vs semi-hard
        float lim = sqrtf(pd) + th;                    // dist < pos_d + thresh
        thr2[row] = lim * lim;                         // compare in squared space
    }
}

// ---------------------------------------------------------------------------
// Kernel 2: fused Gram GEMM (bf16 WMMA) + semi-hard mining.
// A operand = -2e, C initialized with (rA + cA)  =>  accumulator IS d^2.
// 64 blocks x 256 threads; TDM double-buffers 64-col panels into LDS.
// ---------------------------------------------------------------------------
__global__ __launch_bounds__(256)
void mine_kernel(const unsigned short* __restrict__ ebf,
                 const unsigned short* __restrict__ am2,
                 const float* __restrict__ sq, const float* __restrict__ srow,
                 const float* __restrict__ thr2,
                 const int* __restrict__ ids,
                 int* __restrict__ sel, float* __restrict__ valid) {
    __shared__ unsigned short panel[2][64 * D_K];   // 2 x 16 KB, TDM-filled

    const int tid  = threadIdx.x;
    const int wv   = tid >> 5;
    const int lane = tid & 31;
    const int m    = lane & 15;     // row-in-tile for A, col-in-tile for B/C
    const int h    = lane >> 4;     // half-wave select

    const int waveRow = blockIdx.x * 128 + wv * 16;
    const int myRow0  = waveRow + h * 8;   // C layout: VGPR r -> row r + 8*h

    // Kick off TDM for panel 0 (wave 0 only; EXEC-independent DMA)
    if (tid < 32) tdm_panel_load(ebf, (unsigned)(uintptr_t)&panel[0][0]);

    // --- A fragments (-2e): 16x32 bf16 per K-step, ISA A-layout, 2x B128 ---
    Frag A[4];
    {
        const unsigned short* rowp = am2 + (size_t)(waveRow + m) * D_K;
        #pragma unroll
        for (int kk = 0; kk < 4; ++kk) {
            A[kk].q.lo = *(const U4*)(rowp + kk * 32 + h * 8);
            A[kk].q.hi = *(const U4*)(rowp + kk * 32 + 16 + h * 8);
        }
    }

    // --- per-row constants for the 8 rows this lane covers in C ---
    float rA[8], rt2[8];
    int   rid[8];
    unsigned rha[8];
    const float cEps  = 2.0f * EPSV;
    const float dEps2 = (float)D_K * EPSV * EPSV;
    #pragma unroll
    for (int r = 0; r < 8; ++r) {
        const int i = myRow0 + r;
        rA[r]  = sq[i] + cEps * srow[i] + dEps2;   // row part of d^2
        rt2[r] = thr2[i];
        rid[r] = ids[i];
        rha[r] = (unsigned)i * 0x9E3779B1u + 0x9D2C5681u;  // row hash term
    }

    unsigned besth[8];
    float    mind[8];
    int      bestj[8], minj[8];
    #pragma unroll
    for (int r = 0; r < 8; ++r) { besth[r] = 0u; mind[r] = BIGF; bestj[r] = 0; minj[r] = 0; }

    __builtin_amdgcn_s_wait_tensorcnt(0);
    __syncthreads();

    for (int p = 0; p < B_N / 64; ++p) {
        const int col0 = p * 64;
        // prefetch next panel via TDM while computing this one
        if (p + 1 < B_N / 64 && tid < 32)
            tdm_panel_load(ebf + (size_t)(col0 + 64) * D_K,
                           (unsigned)(uintptr_t)&panel[(p + 1) & 1][0]);

        const unsigned short* pan = &panel[p & 1][0];

        #pragma unroll
        for (int sub = 0; sub < 4; ++sub) {
            const int   col = col0 + sub * 16 + m;
            const float cA  = sq[col] - cEps * srow[col];   // col part of d^2
            const int   cid = ids[col];
            const unsigned ch = (unsigned)col * 0x85EBCA77u;

            // C init: rA + cA; A = -2e  =>  final acc[r] = d^2 exactly
            v8f acc;
            #pragma unroll
            for (int r = 0; r < 8; ++r) acc[r] = rA[r] + cA;

            const unsigned short* colp = pan + (sub * 16 + m) * D_K;
            #pragma unroll
            for (int kk = 0; kk < 4; ++kk) {
                Frag Bf;
                Bf.q.lo = *(const U4*)(colp + kk * 32 + h * 8);       // ds_load_b128
                Bf.q.hi = *(const U4*)(colp + kk * 32 + 16 + h * 8);
                acc = __builtin_amdgcn_wmma_f32_16x16x32_bf16(
                    false, A[kk].v, false, Bf.v, (short)0, acc, false, false);
            }

            #pragma unroll
            for (int r = 0; r < 8; ++r) {
                const float d2  = acc[r];
                const bool  neg = (rid[r] != cid);
                const bool  semi = neg && (d2 < rt2[r]);     // squared-space test
                // Gumbel-argmax == argmax of per-pair uniform hash (monotone map)
                unsigned hh = (rha[r] ^ ch) * 0x2C1B3C6Du;
                hh ^= hh >> 16;
                const unsigned hs = semi ? hh : 0u;          // v_cndmask
                const bool takeS = hs > besth[r];
                besth[r] = (besth[r] >= hs) ? besth[r] : hs; // v_max_u32
                bestj[r] = takeS ? col : bestj[r];           // v_cndmask
                const float dn = neg ? d2 : BIGF;            // v_cndmask
                const bool takeM = dn < mind[r];
                mind[r] = fminf(mind[r], dn);                // v_min_num_f32
                minj[r] = takeM ? col : minj[r];             // v_cndmask
            }
        }

        __builtin_amdgcn_s_wait_tensorcnt(0);
        __syncthreads();
    }

    // reduce across the 16 lanes sharing each C row (xor masks stay within half)
    #pragma unroll
    for (int r = 0; r < 8; ++r) {
        #pragma unroll
        for (int off = 8; off >= 1; off >>= 1) {
            unsigned oh = (unsigned)__shfl_xor((int)besth[r], off, 32);
            int      oj = __shfl_xor(bestj[r], off, 32);
            bool tS = oh > besth[r];
            besth[r] = tS ? oh : besth[r];
            bestj[r] = tS ? oj : bestj[r];
            float od = __shfl_xor(mind[r], off, 32);
            int   om = __shfl_xor(minj[r], off, 32);
            bool tM = od < mind[r];
            mind[r] = tM ? od : mind[r];
            minj[r] = tM ? om : minj[r];
        }
    }

    if (m == 0) {
        #pragma unroll
        for (int r = 0; r < 8; ++r) {
            const int  i       = myRow0 + r;
            const bool hasSemi = besth[r] > 0u;
            sel[i]   = hasSemi ? bestj[r] : minj[r];
            valid[i] = (mind[r] < 1.0e38f) ? 1.0f : 0.0f;   // any true negative exists
        }
    }
}

// ---------------------------------------------------------------------------
// Kernel 3: fp32 triplet loss on selected negatives. One wave per row.
// ---------------------------------------------------------------------------
__global__ __launch_bounds__(256)
void loss_kernel(const float* __restrict__ e, const float* __restrict__ pe,
                 const int* __restrict__ sel, const float* __restrict__ valid,
                 float* __restrict__ acc) {
    const int wv   = threadIdx.x >> 5;
    const int lane = threadIdx.x & 31;
    const int row  = blockIdx.x * 8 + wv;
    const int s    = sel[row];

    const float4 ev = *(const float4*)(e  + (size_t)row * D_K + lane * 4);
    const float4 pv = *(const float4*)(pe + (size_t)row * D_K + lane * 4);
    const float4 nv = *(const float4*)(e  + (size_t)s   * D_K + lane * 4);

    float px = ev.x - pv.x + EPSV, py = ev.y - pv.y + EPSV;
    float pz = ev.z - pv.z + EPSV, pw = ev.w - pv.w + EPSV;
    float nx = ev.x - nv.x + EPSV, ny = ev.y - nv.y + EPSV;
    float nz = ev.z - nv.z + EPSV, nw = ev.w - nv.w + EPSV;
    float pd = px*px + py*py + pz*pz + pw*pw;
    float nd = nx*nx + ny*ny + nz*nz + nw*nw;

    #pragma unroll
    for (int off = 16; off >= 1; off >>= 1) {
        pd += __shfl_xor(pd, off, 32);
        nd += __shfl_xor(nd, off, 32);
    }

    if (lane == 0) {
        float tl = fmaxf(sqrtf(pd) - sqrtf(nd) + MARGINV, 0.0f);
        float w  = valid[row];
        atomicAdd(&acc[0], tl * w);
        atomicAdd(&acc[1], w);
    }
}

__global__ void init_kernel(float* acc) { acc[0] = 0.0f; acc[1] = 0.0f; }
__global__ void final_kernel(const float* acc, float* out) { out[0] = acc[0] / acc[1]; }

// ---------------------------------------------------------------------------
extern "C" void kernel_launch(void* const* d_in, const int* in_sizes, int n_in,
                              void* d_out, int out_size, void* d_ws, size_t ws_size,
                              hipStream_t stream) {
    (void)in_sizes; (void)n_in; (void)out_size; (void)ws_size;
    const float* e   = (const float*)d_in[0];
    const float* pe  = (const float*)d_in[1];
    const int*   ids = (const int*)d_in[2];
    float* out = (float*)d_out;

    char* ws = (char*)d_ws;
    size_t off = 0;
    unsigned short* ebf = (unsigned short*)(ws + off); off += (size_t)B_N * D_K * sizeof(unsigned short);
    unsigned short* am2 = (unsigned short*)(ws + off); off += (size_t)B_N * D_K * sizeof(unsigned short);
    float* sq     = (float*)(ws + off); off += B_N * sizeof(float);
    float* srow   = (float*)(ws + off); off += B_N * sizeof(float);
    float* thr2   = (float*)(ws + off); off += B_N * sizeof(float);
    float* valid  = (float*)(ws + off); off += B_N * sizeof(float);
    int*   sel    = (int*)  (ws + off); off += B_N * sizeof(int);
    float* acc    = (float*)(ws + off); off += 2 * sizeof(float);

    init_kernel<<<1, 1, 0, stream>>>(acc);
    prep_kernel<<<B_N / 8, 256, 0, stream>>>(e, pe, ebf, am2, sq, srow, thr2);
    mine_kernel<<<B_N / 128, 256, 0, stream>>>(ebf, am2, sq, srow, thr2, ids, sel, valid);
    loss_kernel<<<B_N / 8, 256, 0, stream>>>(e, pe, sel, valid, acc);
    final_kernel<<<1, 1, 0, stream>>>(acc, out);
}
